// FeatureAlign_5549097747022
// MI455X (gfx1250) — matrix operand order repaired and checked by reading, not compiled, not run on hardware
//
#include <hip/hip_runtime.h>
#include <hip/hip_bf16.h>

// ---------------------------------------------------------------------------
// FeatureAlign on MI455X (gfx1250): bf16 WMMA GEMMs + fused deformable conv.
// Round 2: N=32 pixel tiles for the K=1152 GEMMs (2 WMMAs per A-load).
// ---------------------------------------------------------------------------

typedef __attribute__((ext_vector_type(16))) __bf16 v16bf;
typedef __attribute__((ext_vector_type(8)))  float  v8f;

struct alignas(16) Q4 { unsigned int x[4]; };
union ABits { Q4 q[2]; v16bf v; };

#define CC   128
#define HH   128
#define WW   128
#define HWSZ 16384
#define BB   2
#define GG   8
#define KKT  9          // 3x3 taps

static __device__ inline unsigned short f2bf(float f) {
    unsigned int u = __float_as_uint(f);
    unsigned int r = (u + 0x7FFFu + ((u >> 16) & 1u)) >> 16;
    return (unsigned short)r;
}

static __device__ inline float sigmoidf(float x) {
    return 1.f / (1.f + __expf(-x));
}

// ---------------------------------------------------------------------------
// 1) Global average pool: pooled[b*C+c] = mean(feat_l[b,c,:,:])
// ---------------------------------------------------------------------------
__global__ void pool_kernel(const float* __restrict__ feat_l, float* __restrict__ pooled) {
    int bc = blockIdx.x;                       // 0 .. B*C-1
    const float* p = feat_l + (size_t)bc * HWSZ;
    float s = 0.f;
    for (int i = threadIdx.x; i < HWSZ; i += 256) s += p[i];
    __shared__ float red[256];
    red[threadIdx.x] = s;
    __syncthreads();
    for (int o = 128; o > 0; o >>= 1) {
        if (threadIdx.x < o) red[threadIdx.x] += red[threadIdx.x + o];
        __syncthreads();
    }
    if (threadIdx.x == 0) pooled[bc] = red[0] * (1.f / HWSZ);
}

// ---------------------------------------------------------------------------
// 2) atten[b,o] = sigmoid(relu(sum_c w_atten[o,c]*pooled[b,c]))
// ---------------------------------------------------------------------------
__global__ void atten_kernel(const float* __restrict__ w_atten,
                             const float* __restrict__ pooled,
                             float* __restrict__ atten) {
    int b = blockIdx.x, o = threadIdx.x;
    const float* wr = w_atten + (size_t)o * CC;
    const float* pb = pooled + (size_t)b * CC;
    float s = 0.f;
    for (int c = 0; c < CC; ++c) s += wr[c] * pb[c];
    atten[b * CC + o] = sigmoidf(fmaxf(s, 0.f));
}

// ---------------------------------------------------------------------------
// 3) bilinear upsample 64x64 -> 128x128 (half-pixel, edge clamp)
// ---------------------------------------------------------------------------
__global__ void upsample_kernel(const float* __restrict__ feat_s, float* __restrict__ feat_up) {
    size_t idx = (size_t)blockIdx.x * 256 + threadIdx.x;   // over B*C*HW
    int x = idx & 127;
    int y = (idx >> 7) & 127;
    size_t bc = idx >> 14;
    float sy = fminf(fmaxf(y * 0.5f - 0.25f, 0.f), 63.f);
    float sx = fminf(fmaxf(x * 0.5f - 0.25f, 0.f), 63.f);
    int y0 = (int)sy, x0 = (int)sx;
    int y1 = min(y0 + 1, 63), x1 = min(x0 + 1, 63);
    float fy = sy - (float)y0, fx = sx - (float)x0;
    const float* s = feat_s + bc * 4096;
    float v = (1.f - fy) * ((1.f - fx) * s[y0 * 64 + x0] + fx * s[y0 * 64 + x1]) +
              fy * ((1.f - fx) * s[y1 * 64 + x0] + fx * s[y1 * 64 + x1]);
    feat_up[idx] = v;
}

// ---------------------------------------------------------------------------
// 4) f32 -> bf16 weight conversion
// ---------------------------------------------------------------------------
__global__ void cvt_bf16_kernel(const float* __restrict__ src, unsigned short* __restrict__ dst, int n) {
    int i = blockIdx.x * 256 + threadIdx.x;
    if (i < n) dst[i] = f2bf(src[i]);
}

// ---------------------------------------------------------------------------
// WMMA helpers. A row-major MxK bf16 (global); B transposed in LDS: ldsB[n][k].
// ---------------------------------------------------------------------------
static __device__ inline v16bf load_a(const unsigned short* __restrict__ arow_base) {
    ABits a;
    a.q[0] = *(const Q4*)(arow_base);
    a.q[1] = *(const Q4*)(arow_base + 8);
    return a.v;
}
static __device__ inline v16bf load_b(const unsigned short* __restrict__ brow_base) {
    ABits b;
    b.q[0] = *(const Q4*)(brow_base);
    b.q[1] = *(const Q4*)(brow_base + 8);
    return b.v;
}
static __device__ inline v8f wmma_acc(v16bf a, v16bf b, v8f acc) {
    return __builtin_amdgcn_wmma_f32_16x16x32_bf16(false, a, false, b,
                                                   (short)0, acc, false, false);
}

// ---------------------------------------------------------------------------
// 5) feat_arm = relu(W_fsm (128x128) * x),  x[c,p] = feat_l*(1+atten)
//    one block = 16-pixel tile; 8 waves = 8 M-tiles.
// ---------------------------------------------------------------------------
__global__ void gemm_fsm_kernel(const float* __restrict__ feat_l,
                                const float* __restrict__ atten,
                                const unsigned short* __restrict__ wfsm,
                                unsigned short* __restrict__ feat_arm) {
    constexpr int K = CC;                      // 128
    __shared__ __align__(16) unsigned short ldsB[16][K + 8];
    int b  = blockIdx.x >> 10;                 // 1024 tiles per batch
    int p0 = (blockIdx.x & 1023) << 4;

    int n  = threadIdx.x & 15;
    int kr = threadIdx.x >> 4;
    const float* fb = feat_l + (size_t)b * CC * HWSZ;
    const float* at = atten + (size_t)b * CC;
    for (int j = 0; j < K; j += 16) {
        int k = kr + j;
        float v = fb[(size_t)k * HWSZ + p0 + n] * (1.f + at[k]);
        ldsB[n][k] = f2bf(v);
    }
    __syncthreads();

    int lane = threadIdx.x & 31, wave = threadIdx.x >> 5;
    int nl = lane & 15, half = lane >> 4;
    int m0 = wave * 16;
    v8f acc = {};
    for (int k0 = 0; k0 < K; k0 += 32) {
        v16bf a = load_a(wfsm + (size_t)(m0 + nl) * K + k0 + half * 16);
        v16bf bm = load_b(&ldsB[nl][k0 + half * 16]);
        acc = wmma_acc(a, bm, acc);
    }
    unsigned short* out = feat_arm + (size_t)b * CC * HWSZ;
#pragma unroll
    for (int v = 0; v < 8; ++v) {
        int row = m0 + half * 8 + v;
        out[(size_t)row * HWSZ + p0 + nl] = f2bf(fmaxf(acc[v], 0.f));
    }
}

// ---------------------------------------------------------------------------
// 6) offset_feat = relu(W_off (128x256) * [feat_arm ; 2*feat_up])
// ---------------------------------------------------------------------------
__global__ void gemm_off_kernel(const unsigned short* __restrict__ feat_arm,
                                const float* __restrict__ feat_up,
                                const unsigned short* __restrict__ woff,
                                unsigned short* __restrict__ offset_feat) {
    constexpr int K = 2 * CC;                  // 256
    __shared__ __align__(16) unsigned short ldsB[16][K + 8];
    int b  = blockIdx.x >> 10;
    int p0 = (blockIdx.x & 1023) << 4;

    int n  = threadIdx.x & 15;
    int kr = threadIdx.x >> 4;
    for (int j = 0; j < K; j += 16) {
        int k = kr + j;
        unsigned short v;
        if (k < CC) v = feat_arm[(size_t)b * CC * HWSZ + (size_t)k * HWSZ + p0 + n];
        else        v = f2bf(2.f * feat_up[(size_t)b * CC * HWSZ + (size_t)(k - CC) * HWSZ + p0 + n]);
        ldsB[n][k] = v;
    }
    __syncthreads();

    int lane = threadIdx.x & 31, wave = threadIdx.x >> 5;
    int nl = lane & 15, half = lane >> 4;
    int m0 = wave * 16;
    v8f acc = {};
    for (int k0 = 0; k0 < K; k0 += 32) {
        v16bf a = load_a(woff + (size_t)(m0 + nl) * K + k0 + half * 16);
        v16bf bm = load_b(&ldsB[nl][k0 + half * 16]);
        acc = wmma_acc(a, bm, acc);
    }
    unsigned short* out = offset_feat + (size_t)b * CC * HWSZ;
#pragma unroll
    for (int v = 0; v < 8; ++v) {
        int row = m0 + half * 8 + v;
        out[(size_t)row * HWSZ + p0 + nl] = f2bf(fmaxf(acc[v], 0.f));
    }
}

// ---------------------------------------------------------------------------
// 7) om = conv3x3(offset_feat, w_offmask) + b_offmask
//    implicit GEMM: M=216 (14 tiles), K=1152, N=32-pixel tile.
//    Each A-row b128 pair feeds 2 WMMAs (two N-subtiles).
// ---------------------------------------------------------------------------
__global__ void conv_offmask_kernel(const unsigned short* __restrict__ offset_feat,
                                    const unsigned short* __restrict__ womask,
                                    const float* __restrict__ b_offmask,
                                    float* __restrict__ om) {
    constexpr int K = CC * KKT;                // 1152
    constexpr int M = 3 * GG * KKT;            // 216
    __shared__ __align__(16) unsigned short ldsB[32][K + 8];
    int b  = blockIdx.x >> 9;                  // 512 tiles of 32 px per batch
    int p0 = (blockIdx.x & 511) << 5;
    int y  = p0 >> 7;
    int x0 = p0 & 127;                         // 0,32,64,96 -> tile never crosses a row

    const unsigned short* in = offset_feat + (size_t)b * CC * HWSZ;
    for (int t = threadIdx.x; t < 32 * K; t += 256) {
        int n = t & 31;
        int k = t >> 5;
        int c = k / 9;
        int r = k - c * 9;
        int yy = y + (r / 3) - 1;
        int xx = x0 + n + (r % 3) - 1;
        unsigned short v = 0;
        if (yy >= 0 && yy < HH && xx >= 0 && xx < WW)
            v = in[(size_t)c * HWSZ + yy * WW + xx];
        ldsB[n][k] = v;
    }
    __syncthreads();

    int lane = threadIdx.x & 31, wave = threadIdx.x >> 5;
    int nl = lane & 15, half = lane >> 4;
    bool two = (wave < 6);                     // tiles 8..13 for waves 0..5
    int mA = wave * 16;
    int mB = (wave + 8) * 16;
    v8f a00 = {}, a01 = {}, a10 = {}, a11 = {};
    for (int k0 = 0; k0 < K; k0 += 32) {
        v16bf b0 = load_b(&ldsB[nl][k0 + half * 16]);
        v16bf b1 = load_b(&ldsB[nl + 16][k0 + half * 16]);
        const unsigned short* arow0 = womask + (size_t)(mA + nl) * K + k0 + half * 16;
        if (k0 + 32 < K) __builtin_prefetch(arow0 + 32, 0, 0);
        v16bf av0 = load_a(arow0);
        a00 = wmma_acc(av0, b0, a00);
        a01 = wmma_acc(av0, b1, a01);
        if (two) {
            v16bf av1 = load_a(womask + (size_t)(mB + nl) * K + k0 + half * 16);
            a10 = wmma_acc(av1, b0, a10);
            a11 = wmma_acc(av1, b1, a11);
        }
    }
    float* ob = om + (size_t)b * M * HWSZ;
#pragma unroll
    for (int v = 0; v < 8; ++v) {
        int row = mA + half * 8 + v;
        float bias = b_offmask[row];
        ob[(size_t)row * HWSZ + p0 + nl]      = a00[v] + bias;
        ob[(size_t)row * HWSZ + p0 + 16 + nl] = a01[v] + bias;
    }
    if (two) {
#pragma unroll
        for (int v = 0; v < 8; ++v) {
            int row = mB + half * 8 + v;
            if (row < M) {
                float bias = b_offmask[row];
                ob[(size_t)row * HWSZ + p0 + nl]      = a10[v] + bias;
                ob[(size_t)row * HWSZ + p0 + 16 + nl] = a11[v] + bias;
            }
        }
    }
}

// ---------------------------------------------------------------------------
// 8) Modulated deformable conv + relu + residual (fused sampling + WMMA GEMM)
//    K = G*Cg*KK = 1152 (k = g*144 + c*9 + kk, matching w_dcn (128,1152) rows)
//    N=32 pixel tile; each A-load feeds 2 WMMAs.
// ---------------------------------------------------------------------------
__global__ void dcn_kernel(const float* __restrict__ feat_up,
                           const float* __restrict__ om,
                           const unsigned short* __restrict__ wdcn,
                           const float* __restrict__ b_dcn,
                           const float* __restrict__ feat_l,
                           float* __restrict__ out) {
    constexpr int K = CC * KKT;                // 1152
    constexpr int M = 3 * GG * KKT;            // 216 om channels
    __shared__ __align__(16) unsigned short ldsB[32][K + 8];
    int b  = blockIdx.x >> 9;
    int p0 = (blockIdx.x & 511) << 5;
    int y  = p0 >> 7;
    int x0 = p0 & 127;

    const float* omb = om + (size_t)b * M * HWSZ;
    const float* Fb  = feat_up + (size_t)b * CC * HWSZ;

    // sample phase: 72 (g,kk) points x 32 pixels; corners reused across 16 ch
    for (int t = threadIdx.x; t < 72 * 32; t += 256) {
        int n  = t & 31;
        int gk = t >> 5;                       // 0..71
        int g  = gk / 9;
        int kk = gk - g * 9;
        int pix = p0 + n;
        float dy = omb[(size_t)(g * 18 + kk * 2)     * HWSZ + pix];
        float dx = omb[(size_t)(g * 18 + kk * 2 + 1) * HWSZ + pix];
        float ms = sigmoidf(omb[(size_t)(144 + g * 9 + kk) * HWSZ + pix]);
        float py = (float)y        + (float)(kk / 3 - 1) + dy;
        float px = (float)(x0 + n) + (float)(kk % 3 - 1) + dx;
        float fy0 = floorf(py), fx0 = floorf(px);
        int y0 = (int)fy0, x0i = (int)fx0;
        float ly = py - fy0, lx = px - fx0;
        int y1 = y0 + 1, x1 = x0i + 1;
        float vy0 = (y0 >= 0 && y0 < HH) ? 1.f : 0.f;
        float vy1 = (y1 >= 0 && y1 < HH) ? 1.f : 0.f;
        float vx0 = (x0i >= 0 && x0i < WW) ? 1.f : 0.f;
        float vx1 = (x1 >= 0 && x1 < WW) ? 1.f : 0.f;
        float w00 = (1.f - ly) * (1.f - lx) * vy0 * vx0 * ms;
        float w01 = (1.f - ly) * lx * vy0 * vx1 * ms;
        float w10 = ly * (1.f - lx) * vy1 * vx0 * ms;
        float w11 = ly * lx * vy1 * vx1 * ms;
        int iy0 = min(max(y0, 0), HH - 1), iy1 = min(max(y1, 0), HH - 1);
        int ix0 = min(max(x0i, 0), WW - 1), ix1 = min(max(x1, 0), WW - 1);
        int i00 = iy0 * WW + ix0, i01 = iy0 * WW + ix1;
        int i10 = iy1 * WW + ix0, i11 = iy1 * WW + ix1;
        const float* F = Fb + (size_t)(g * 16) * HWSZ;
        int kbase = g * 144 + kk;
#pragma unroll 4
        for (int c = 0; c < 16; ++c) {
            const float* Fc = F + (size_t)c * HWSZ;
            float v = w00 * Fc[i00] + w01 * Fc[i01] + w10 * Fc[i10] + w11 * Fc[i11];
            ldsB[n][kbase + c * 9] = f2bf(v);
        }
    }
    __syncthreads();

    int lane = threadIdx.x & 31, wave = threadIdx.x >> 5;
    int nl = lane & 15, half = lane >> 4;
    int m0 = wave * 16;
    v8f acc0 = {}, acc1 = {};
    for (int k0 = 0; k0 < K; k0 += 32) {
        v16bf b0 = load_b(&ldsB[nl][k0 + half * 16]);
        v16bf b1 = load_b(&ldsB[nl + 16][k0 + half * 16]);
        const unsigned short* arow = wdcn + (size_t)(m0 + nl) * K + k0 + half * 16;
        if (k0 + 32 < K) __builtin_prefetch(arow + 32, 0, 0);
        v16bf av = load_a(arow);
        acc0 = wmma_acc(av, b0, acc0);
        acc1 = wmma_acc(av, b1, acc1);
    }
    const float* resid = feat_l + (size_t)b * CC * HWSZ;
    float* ob = out + (size_t)b * CC * HWSZ;
#pragma unroll
    for (int v = 0; v < 8; ++v) {
        int row = m0 + half * 8 + v;
        float bias = b_dcn[row];
        size_t o0 = (size_t)row * HWSZ + p0 + nl;
        size_t o1 = o0 + 16;
        ob[o0] = fmaxf(acc0[v] + bias, 0.f) + resid[o0];
        ob[o1] = fmaxf(acc1[v] + bias, 0.f) + resid[o1];
    }
}

// ---------------------------------------------------------------------------
// Host launcher
// ---------------------------------------------------------------------------
extern "C" void kernel_launch(void* const* d_in, const int* in_sizes, int n_in,
                              void* d_out, int out_size, void* d_ws, size_t ws_size,
                              hipStream_t stream) {
    const float* feat_l    = (const float*)d_in[0];
    const float* feat_s    = (const float*)d_in[1];
    const float* w_atten   = (const float*)d_in[2];
    const float* w_fsm     = (const float*)d_in[3];
    const float* w_off     = (const float*)d_in[4];
    const float* w_offmask = (const float*)d_in[5];
    const float* b_offmask = (const float*)d_in[6];
    const float* w_dcn     = (const float*)d_in[7];
    const float* b_dcn     = (const float*)d_in[8];
    float* out = (float*)d_out;

    size_t off = 0;
    auto take = [&](size_t bytes) -> void* {
        void* p = (char*)d_ws + off;
        off += (bytes + 255) & ~(size_t)255;
        return p;
    };
    float* pooled            = (float*)take((size_t)BB * CC * 4);
    float* atten             = (float*)take((size_t)BB * CC * 4);
    float* feat_up           = (float*)take((size_t)BB * CC * HWSZ * 4);
    unsigned short* feat_arm = (unsigned short*)take((size_t)BB * CC * HWSZ * 2);
    unsigned short* off_feat = (unsigned short*)take((size_t)BB * CC * HWSZ * 2);
    float* om                = (float*)take((size_t)BB * 216 * HWSZ * 4);
    unsigned short* wfsm_bf  = (unsigned short*)take((size_t)CC * CC * 2);
    unsigned short* woff_bf  = (unsigned short*)take((size_t)CC * 2 * CC * 2);
    unsigned short* womask_bf = (unsigned short*)take((size_t)224 * 1152 * 2); // padded to 224 rows
    unsigned short* wdcn_bf  = (unsigned short*)take((size_t)CC * 1152 * 2);

    // pool + attention
    pool_kernel<<<BB * CC, 256, 0, stream>>>(feat_l, pooled);
    atten_kernel<<<BB, CC, 0, stream>>>(w_atten, pooled, atten);

    // upsample
    upsample_kernel<<<(BB * CC * HWSZ) / 256, 256, 0, stream>>>(feat_s, feat_up);

    // weight conversion to bf16
    cvt_bf16_kernel<<<(CC * CC + 255) / 256, 256, 0, stream>>>(w_fsm, wfsm_bf, CC * CC);
    cvt_bf16_kernel<<<(CC * 2 * CC + 255) / 256, 256, 0, stream>>>(w_off, woff_bf, CC * 2 * CC);
    cvt_bf16_kernel<<<(216 * 1152 + 255) / 256, 256, 0, stream>>>(w_offmask, womask_bf, 216 * 1152);
    cvt_bf16_kernel<<<(CC * 1152 + 255) / 256, 256, 0, stream>>>(w_dcn, wdcn_bf, CC * 1152);

    const int tiles16 = BB * (HWSZ / 16);      // 2048 blocks (N=16)
    const int tiles32 = BB * (HWSZ / 32);      // 1024 blocks (N=32)
    gemm_fsm_kernel<<<tiles16, 256, 0, stream>>>(feat_l, atten, wfsm_bf, feat_arm);
    gemm_off_kernel<<<tiles16, 256, 0, stream>>>(feat_arm, feat_up, woff_bf, off_feat);
    conv_offmask_kernel<<<tiles32, 256, 0, stream>>>(off_feat, womask_bf, b_offmask, om);
    dcn_kernel<<<tiles32, 256, 0, stream>>>(feat_up, om, wdcn_bf, b_dcn, feat_l, out);

    (void)in_sizes; (void)n_in; (void)out_size; (void)ws_size;
}
